// DeepSeekSparseAttention_19542101197224
// MI455X (gfx1250) — matrix operand order repaired and checked by reading, not compiled, not run on hardware
//
#include <hip/hip_runtime.h>
#include <hip/hip_bf16.h>
#include <math.h>

// ---------------- problem constants (match reference) ----------------
#define TT    1024
#define HID   2048
#define NH    16
#define NKV   4
#define DH    128
#define GQ    (NH / NKV)      // 4
#define TOPK  256
#define HI_   4
#define DI_   32

typedef __attribute__((ext_vector_type(16))) _Float16 v16h;
typedef __attribute__((ext_vector_type(8)))  _Float16 h8;
typedef __attribute__((ext_vector_type(8)))  float    v8f;

union FragA { h8 c[2]; v16h v; };

// =====================================================================
// 0a) f32 -> f16 convert (row-major copy)
// =====================================================================
__global__ __launch_bounds__(256)
void dssa_cvt_f16(const float* __restrict__ in, _Float16* __restrict__ out, int n) {
  int i = blockIdx.x * 256 + threadIdx.x;
  if (i < n) out[i] = (_Float16)in[i];
}

// =====================================================================
// 0b) f32 (K x N) -> f16 transposed (N x K)
// =====================================================================
__global__ __launch_bounds__(256)
void dssa_cvt_f16_T(const float* __restrict__ in, _Float16* __restrict__ out,
                    int K, int N) {
  int i = blockIdx.x * 256 + threadIdx.x;
  if (i < K * N) {
    int k = i / N, n = i - k * N;          // coalesced read along n
    out[(size_t)n * K + k] = (_Float16)in[i];
  }
}

// =====================================================================
// 1) Blocked WMMA GEMM:  C[M,N] = A[M,K] * B[K,N]
//    A: f16 row-major (M x K); BT: f16 transposed (N x K); C: f32.
//    Block = 256 threads (8 wave32), C-tile 128(M) x 128(N).
//    K staged in 32-wide slabs through DOUBLE-BUFFERED LDS; the next
//    slab's global_load_b128s are issued before the current slab's 8
//    WMMAs so HBM/L2 latency overlaps the matrix pipe.
//    Per wave: 2 M-tiles x 4 N-tiles = 8 v_wmma per slab.
// =====================================================================
__global__ __launch_bounds__(256)
void dssa_wmma_gemm_big(const _Float16* __restrict__ A,
                        const _Float16* __restrict__ BT,
                        float* __restrict__ C, int M, int N, int K) {
  __shared__ __align__(16) _Float16 As[2][128 * 32];   // 2 x 8KB
  __shared__ __align__(16) _Float16 Bs[2][128 * 32];   // 2 x 8KB

  const int tid  = threadIdx.x;
  const int lane = tid & 31;
  const int l15  = lane & 15;
  const int half = lane >> 4;
  const int w    = tid >> 5;            // wave 0..7
  const int mrow = (w & 3) * 32;        // wave M offset in block tile
  const int ncol = (w >> 2) * 64;       // wave N offset in block tile
  const int m0   = blockIdx.y * 128;
  const int n0   = blockIdx.x * 128;

  // staging assignment: thread owns rows r0 and r0+64, one h8 chunk each
  const int r0  = tid >> 2;
  const int cs  = (tid & 3) * 8;
  const _Float16* Ap0 = A  + (size_t)(m0 + r0)      * K + cs;
  const _Float16* Ap1 = A  + (size_t)(m0 + r0 + 64) * K + cs;
  const _Float16* Bp0 = BT + (size_t)(n0 + r0)      * K + cs;
  const _Float16* Bp1 = BT + (size_t)(n0 + r0 + 64) * K + cs;
  const int dA0 = r0 * 32 + cs;
  const int dA1 = (r0 + 64) * 32 + cs;

  v8f acc00 = {}, acc01 = {}, acc02 = {}, acc03 = {};
  v8f acc10 = {}, acc11 = {}, acc12 = {}, acc13 = {};

  // preload slab 0 into buffer 0
  {
    h8 ra0 = *(const h8*)Ap0, ra1 = *(const h8*)Ap1;
    h8 rb0 = *(const h8*)Bp0, rb1 = *(const h8*)Bp1;
    *(h8*)(&As[0][dA0]) = ra0;  *(h8*)(&As[0][dA1]) = ra1;
    *(h8*)(&Bs[0][dA0]) = rb0;  *(h8*)(&Bs[0][dA1]) = rb1;
  }

  const int nslab = K / 32;
  int buf = 0;
  for (int s = 0; s < nslab; ++s) {
    __syncthreads();

    // ---- issue next slab's global loads (overlap with WMMAs below) ----
    const bool more = (s + 1 < nslab);
    h8 na0, na1, nb0, nb1;
    if (more) {
      int kk = (s + 1) * 32;
      na0 = *(const h8*)(Ap0 + kk);
      na1 = *(const h8*)(Ap1 + kk);
      nb0 = *(const h8*)(Bp0 + kk);
      nb1 = *(const h8*)(Bp1 + kk);
    }

    // ---- A fragments for this wave's two M-tiles ----
    const _Float16* as = As[buf];
    const _Float16* bs = Bs[buf];
    FragA a0, a1;
    {
      const _Float16* ap0 = as + (mrow + l15) * 32;
      const _Float16* ap1 = as + (mrow + 16 + l15) * 32;
      a0.c[0] = *(const h8*)(ap0 + half * 8);
      a0.c[1] = *(const h8*)(ap0 + 16 + half * 8);
      a1.c[0] = *(const h8*)(ap1 + half * 8);
      a1.c[1] = *(const h8*)(ap1 + 16 + half * 8);
    }

#define DSSA_STEP(J, ACCA, ACCB)                                              \
    {                                                                         \
      FragA b;                                                                \
      const _Float16* bp = bs + (ncol + (J) * 16 + l15) * 32;                 \
      b.c[0] = *(const h8*)(bp + half * 8);                                   \
      b.c[1] = *(const h8*)(bp + 16 + half * 8);                              \
      ACCA = __builtin_amdgcn_wmma_f32_16x16x32_f16(                          \
          false, a0.v, false, b.v, (short)0, ACCA, false, false);             \
      ACCB = __builtin_amdgcn_wmma_f32_16x16x32_f16(                          \
          false, a1.v, false, b.v, (short)0, ACCB, false, false);             \
    }
    DSSA_STEP(0, acc00, acc10)
    DSSA_STEP(1, acc01, acc11)
    DSSA_STEP(2, acc02, acc12)
    DSSA_STEP(3, acc03, acc13)
#undef DSSA_STEP

    // ---- commit next slab into the other buffer ----
    if (more) {
      int nb = buf ^ 1;
      *(h8*)(&As[nb][dA0]) = na0;  *(h8*)(&As[nb][dA1]) = na1;
      *(h8*)(&Bs[nb][dA0]) = nb0;  *(h8*)(&Bs[nb][dA1]) = nb1;
    }
    buf ^= 1;
  }

  // ---- store: VGPR r -> row M = r + half*8, col = l15 ----
#define DSSA_STORE(MI, J, ACC)                                                \
  {                                                                           \
    float* Cp = C + (size_t)(m0 + mrow + (MI) * 16 + half * 8) * N +          \
                n0 + ncol + (J) * 16 + l15;                                   \
    _Pragma("unroll")                                                         \
    for (int r = 0; r < 8; ++r) Cp[(size_t)r * N] = ACC[r];                   \
  }
  DSSA_STORE(0, 0, acc00) DSSA_STORE(0, 1, acc01)
  DSSA_STORE(0, 2, acc02) DSSA_STORE(0, 3, acc03)
  DSSA_STORE(1, 0, acc10) DSSA_STORE(1, 1, acc11)
  DSSA_STORE(1, 2, acc12) DSSA_STORE(1, 3, acc13)
#undef DSSA_STORE
}

// =====================================================================
// 1b) Simple one-wave WMMA GEMM (f32 in, converts on load) for narrow N
// =====================================================================
__global__ __launch_bounds__(32)
void dssa_wmma_gemm(const float* __restrict__ A, const float* __restrict__ B,
                    float* __restrict__ C, int M, int N, int K) {
  const int lane = threadIdx.x & 31;
  const int l15  = lane & 15;
  const int half = lane >> 4;
  const int mt   = blockIdx.y;
  const int nt   = blockIdx.x;
  const int arow = mt * 16 + l15;
  const int bcol = nt * 16 + l15;

  v8f acc = {};
  for (int kk = 0; kk < K; kk += 32) {
    const float* Ap = A + (size_t)arow * K + kk;
    const float* Bp = B + (size_t)kk * N + bcol;
    v16h a, b;
#pragma unroll
    for (int v = 0; v < 8; ++v) {
      int ka = ((v < 4) ? 0 : 16) + half * 8 + 2 * (v & 3);
      a[2 * v]     = (_Float16)Ap[ka];
      a[2 * v + 1] = (_Float16)Ap[ka + 1];
      b[2 * v]     = (_Float16)Bp[(size_t)ka * N];
      b[2 * v + 1] = (_Float16)Bp[(size_t)(ka + 1) * N];
    }
    acc = __builtin_amdgcn_wmma_f32_16x16x32_f16(
        false, a, false, b, (short)0, acc, false, false);
  }
  float* Cp = C + (size_t)(mt * 16 + half * 8) * N + nt * 16 + l15;
#pragma unroll
  for (int r = 0; r < 8; ++r) Cp[(size_t)r * N] = acc[r];
}

// =====================================================================
// 2) wi = x @ Wwi   (N = 4)
// =====================================================================
__global__ __launch_bounds__(32)
void dssa_wi(const float* __restrict__ x, const float* __restrict__ Wwi,
             float* __restrict__ wi) {
  const int t = blockIdx.x, lane = threadIdx.x;
  float acc[HI_] = {0.f, 0.f, 0.f, 0.f};
  for (int c = lane; c < HID; c += 32) {
    float xv = x[t * HID + c];
#pragma unroll
    for (int j = 0; j < HI_; ++j) acc[j] += xv * Wwi[c * HI_ + j];
  }
#pragma unroll
  for (int j = 0; j < HI_; ++j) {
    float s = acc[j];
#pragma unroll
    for (int m = 16; m >= 1; m >>= 1) s += __shfl_xor(s, m, 32);
    if (lane == 0) wi[t * HI_ + j] = s;
  }
}

// =====================================================================
// 3) RoPE + RMS-norm in place on q (16 heads) and k (4 heads)
// =====================================================================
__global__ __launch_bounds__(32)
void dssa_rope_rms(float* __restrict__ q, float* __restrict__ k,
                   const float* __restrict__ cosT, const float* __restrict__ sinT) {
  const int t = blockIdx.x;
  const int h = blockIdx.y;
  float* row = (h < NH) ? (q + (size_t)t * (NH * DH) + h * DH)
                        : (k + (size_t)t * (NKV * DH) + (h - NH) * DH);
  __shared__ float xs[DH];
  const int lane = threadIdx.x;
#pragma unroll
  for (int i = 0; i < 4; ++i) xs[lane * 4 + i] = row[lane * 4 + i];
  __syncthreads();

  float y[4];
  float ss = 0.f;
#pragma unroll
  for (int i = 0; i < 4; ++i) {
    int d = lane * 4 + i;
    float c = cosT[t * DH + d], s = sinT[t * DH + d];
    float rot = (d < DH / 2) ? -xs[d + DH / 2] : xs[d - DH / 2];
    y[i] = xs[d] * c + rot * s;
    ss += y[i] * y[i];
  }
#pragma unroll
  for (int m = 16; m >= 1; m >>= 1) ss += __shfl_xor(ss, m, 32);
  float scale = rsqrtf(ss * (1.f / DH) + 1.1920929e-7f);
#pragma unroll
  for (int i = 0; i < 4; ++i) row[lane * 4 + i] = y[i] * scale;
}

// =====================================================================
// 4) Indexer scores: I[t,s] = sum_h wi[t,h]*relu(qi[t,h,:].ki[s,:]), causal
// =====================================================================
__global__ __launch_bounds__(256)
void dssa_indexer(const float* __restrict__ qi, const float* __restrict__ ki,
                  const float* __restrict__ wi, float* __restrict__ scores) {
  const int t = blockIdx.x, tid = threadIdx.x;
  __shared__ float qis[HI_ * DI_];
  __shared__ float wis[HI_];
  if (tid < HI_ * DI_) qis[tid] = qi[t * (HI_ * DI_) + tid];
  if (tid < HI_)       wis[tid] = wi[t * HI_ + tid];
  __syncthreads();
#pragma unroll
  for (int r = 0; r < TT / 256; ++r) {
    int s = tid + r * 256;
    float sc;
    if (s > t) {
      sc = -__builtin_inff();
    } else {
      sc = 0.f;
      const float* kr = ki + s * DI_;
#pragma unroll
      for (int h = 0; h < HI_; ++h) {
        float dot = 0.f;
#pragma unroll
        for (int d = 0; d < DI_; ++d) dot += qis[h * DI_ + d] * kr[d];
        sc += wis[h] * fmaxf(dot, 0.f);
      }
    }
    scores[t * TT + s] = sc;
  }
}

// =====================================================================
// 5) Top-256 per row via in-LDS bitonic sort (descending)
// =====================================================================
__global__ __launch_bounds__(256)
void dssa_topk(const float* __restrict__ scores, int* __restrict__ topI) {
  const int t = blockIdx.x, tid = threadIdx.x;
  __shared__ float sv[TT];
  __shared__ int   si[TT];
#pragma unroll
  for (int r = 0; r < TT / 256; ++r) {
    int s = tid + r * 256;
    sv[s] = scores[t * TT + s];
    si[s] = s;
  }
  for (unsigned k = 2; k <= TT; k <<= 1) {
    for (unsigned j = k >> 1; j > 0; j >>= 1) {
      __syncthreads();
#pragma unroll
      for (int r = 0; r < TT / 256; ++r) {
        unsigned i = tid + r * 256u;
        unsigned ixj = i ^ j;
        if (ixj > i) {
          bool dir = ((i & k) == 0);
          float a = sv[i], b = sv[ixj];
          if ((a < b) == dir) {
            sv[i] = b; sv[ixj] = a;
            int ti = si[i]; si[i] = si[ixj]; si[ixj] = ti;
          }
        }
      }
    }
  }
  __syncthreads();
  topI[t * TOPK + tid] = si[tid];
}

// =====================================================================
// 6) Sparse grouped-query attention; emits f16 for the Wo GEMM.
// =====================================================================
__global__ __launch_bounds__(32)
void dssa_attn(const float* __restrict__ q, const float* __restrict__ k,
               const float* __restrict__ v, const int* __restrict__ topI,
               _Float16* __restrict__ attn_h) {
  const int h = blockIdx.x;
  const int t = blockIdx.y;
  const int n = h / GQ;
  const int lane = threadIdx.x;
  const float scale = 0.08838834764831845f;   // 128^-0.5

  __shared__ float qrow[DH];
  __shared__ float p[TOPK];
  __shared__ int   sidx[TOPK];

#pragma unroll
  for (int i = 0; i < 4; ++i)
    qrow[lane * 4 + i] = q[(size_t)t * (NH * DH) + h * DH + lane * 4 + i];
  __syncthreads();

  float sc[8];
  float mx = -__builtin_inff();
#pragma unroll
  for (int j = 0; j < 8; ++j) {
    int ks = lane + 32 * j;
    int idx = topI[t * TOPK + ks];
    sidx[ks] = idx;
    if (idx <= t) {
      const float* kr = k + (size_t)idx * (NKV * DH) + n * DH;
      float dot = 0.f;
#pragma unroll 8
      for (int d = 0; d < DH; ++d) dot += qrow[d] * kr[d];
      sc[j] = dot * scale;
    } else {
      sc[j] = -__builtin_inff();
    }
    mx = fmaxf(mx, sc[j]);
  }
#pragma unroll
  for (int m = 16; m >= 1; m >>= 1) mx = fmaxf(mx, __shfl_xor(mx, m, 32));

  float sum = 0.f;
#pragma unroll
  for (int j = 0; j < 8; ++j) {
    float e = __expf(sc[j] - mx);
    p[lane + 32 * j] = e;
    sum += e;
  }
#pragma unroll
  for (int m = 16; m >= 1; m >>= 1) sum += __shfl_xor(sum, m, 32);
  float inv = 1.f / sum;
  __syncthreads();

  float acc[4] = {0.f, 0.f, 0.f, 0.f};
  for (int ks = 0; ks < TOPK; ++ks) {
    float w = p[ks];
    const float* vr = v + (size_t)sidx[ks] * (NKV * DH) + n * DH + lane * 4;
#pragma unroll
    for (int i = 0; i < 4; ++i) acc[i] += w * vr[i];
  }
#pragma unroll
  for (int i = 0; i < 4; ++i)
    attn_h[(size_t)t * (NH * DH) + h * DH + lane * 4 + i] = (_Float16)(inv * acc[i]);
}

// =====================================================================
// Launch
// =====================================================================
extern "C" void kernel_launch(void* const* d_in, const int* in_sizes, int n_in,
                              void* d_out, int out_size, void* d_ws, size_t ws_size,
                              hipStream_t stream) {
  const float* x    = (const float*)d_in[0];
  const float* cosT = (const float*)d_in[1];
  const float* sinT = (const float*)d_in[2];
  const float* Wq   = (const float*)d_in[3];
  const float* Wk   = (const float*)d_in[4];
  const float* Wv   = (const float*)d_in[5];
  const float* Wo   = (const float*)d_in[6];
  const float* Wqi  = (const float*)d_in[7];
  const float* Wwi  = (const float*)d_in[8];
  const float* Wki  = (const float*)d_in[9];
  float* out = (float*)d_out;

  // ---------------- workspace layout ----------------
  char* p = (char*)d_ws;
  float* q      = (float*)p; p += (size_t)TT * NH * DH * 4;
  float* k      = (float*)p; p += (size_t)TT * NKV * DH * 4;
  float* v      = (float*)p; p += (size_t)TT * NKV * DH * 4;
  float* qi     = (float*)p; p += (size_t)TT * HI_ * DI_ * 4;
  float* wi     = (float*)p; p += (size_t)TT * HI_ * 4;
  float* ki     = (float*)p; p += (size_t)TT * DI_ * 4;
  float* scores = (float*)p; p += (size_t)TT * TT * 4;
  int*   topI   = (int*)p;   p += (size_t)TT * TOPK * 4;
  _Float16* xh    = (_Float16*)p; p += (size_t)TT * HID * 2;
  _Float16* attnh = (_Float16*)p; p += (size_t)TT * NH * DH * 2;
  _Float16* WqT   = (_Float16*)p; p += (size_t)HID * (NH * DH) * 2;
  _Float16* WkT   = (_Float16*)p; p += (size_t)HID * (NKV * DH) * 2;
  _Float16* WvT   = (_Float16*)p; p += (size_t)HID * (NKV * DH) * 2;
  _Float16* WqiT  = (_Float16*)p; p += (size_t)HID * (HI_ * DI_) * 2;
  _Float16* WoT   = (_Float16*)p; p += (size_t)(NH * DH) * HID * 2;

  dim3 b32(32), b256(256);
  const int nx = TT * HID;

  // ---- one-time f16 conversion / weight transposition ----
  dssa_cvt_f16<<<dim3((nx + 255) / 256), b256, 0, stream>>>(x, xh, nx);
  dssa_cvt_f16_T<<<dim3((HID * NH * DH + 255) / 256), b256, 0, stream>>>(Wq, WqT, HID, NH * DH);
  dssa_cvt_f16_T<<<dim3((HID * NKV * DH + 255) / 256), b256, 0, stream>>>(Wk, WkT, HID, NKV * DH);
  dssa_cvt_f16_T<<<dim3((HID * NKV * DH + 255) / 256), b256, 0, stream>>>(Wv, WvT, HID, NKV * DH);
  dssa_cvt_f16_T<<<dim3((HID * HI_ * DI_ + 255) / 256), b256, 0, stream>>>(Wqi, WqiT, HID, HI_ * DI_);
  dssa_cvt_f16_T<<<dim3((NH * DH * HID + 255) / 256), b256, 0, stream>>>(Wo, WoT, NH * DH, HID);

  // ---- projections (blocked WMMA GEMMs, C = x @ W) ----
  dssa_wmma_gemm_big<<<dim3((NH * DH) / 128, TT / 128), b256, 0, stream>>>(xh, WqT, q, TT, NH * DH, HID);
  dssa_wmma_gemm_big<<<dim3((NKV * DH) / 128, TT / 128), b256, 0, stream>>>(xh, WkT, k, TT, NKV * DH, HID);
  dssa_wmma_gemm_big<<<dim3((NKV * DH) / 128, TT / 128), b256, 0, stream>>>(xh, WvT, v, TT, NKV * DH, HID);
  dssa_wmma_gemm_big<<<dim3((HI_ * DI_) / 128, TT / 128), b256, 0, stream>>>(xh, WqiT, qi, TT, HI_ * DI_, HID);
  dssa_wmma_gemm<<<dim3(DI_ / 16, TT / 16), b32, 0, stream>>>(x, Wki, ki, TT, DI_, HID);  // N=32
  dssa_wi<<<dim3(TT), b32, 0, stream>>>(x, Wwi, wi);                                      // N=4

  // ---- RoPE + RMS-norm ----
  dssa_rope_rms<<<dim3(TT, NH + NKV), b32, 0, stream>>>(q, k, cosT, sinT);

  // ---- lightning indexer + top-k ----
  dssa_indexer<<<dim3(TT), b256, 0, stream>>>(qi, ki, wi, scores);
  dssa_topk<<<dim3(TT), b256, 0, stream>>>(scores, topI);

  // ---- sparse GQA (writes f16 activations) ----
  dssa_attn<<<dim3(NH, TT), b32, 0, stream>>>(q, k, v, topI, attnh);

  // ---- output projection: out = attn @ Wo ----
  dssa_wmma_gemm_big<<<dim3(HID / 128, TT / 128), b256, 0, stream>>>(attnh, WoT, out, TT, HID, HID);
}
